// HRFormerBlock_84765474554700
// MI455X (gfx1250) — matrix operand order repaired and checked by reading, not compiled, not run on hardware
//
#include <hip/hip_runtime.h>
#include <hip/hip_bf16.h>
#include <hip/hip_fp16.h>

// ---------------------------------------------------------------------------
// HRFormer/Swin block for MI455X (gfx1250, wave32, WMMA).
// GEMMs: f16 x f16 -> f32 via v_wmma_f32_16x16x32_f16.
// Data movement: TDM tensor_load_to_lds (A tiles) + global_load_async_to_lds
// (B tiles) when the toolchain exposes the builtins; manual staging otherwise.
// ---------------------------------------------------------------------------

typedef _Float16 h16;
typedef __attribute__((ext_vector_type(16))) _Float16 v16h;
typedef __attribute__((ext_vector_type(8)))  float    v8f;
typedef __attribute__((ext_vector_type(4)))  unsigned v4u;
typedef __attribute__((ext_vector_type(8)))  int      v8i;
typedef __attribute__((ext_vector_type(4)))  int      v4i;

#if __has_builtin(__builtin_amdgcn_global_load_async_to_lds_b128)
#define HAVE_ASYNC 1
#endif
#if __has_builtin(__builtin_amdgcn_tensor_load_to_lds) && \
    __has_builtin(__builtin_amdgcn_s_wait_tensorcnt)
#define HAVE_TDM 1
#endif

union FragA { v16h v; unsigned u[8]; };
union FragC { v8f  v; float    f[8]; };

__device__ __forceinline__ v8f zero8() {
    FragC z;
    #pragma unroll
    for (int e = 0; e < 8; ++e) z.f[e] = 0.f;
    return z.v;
}

#define WMMA_F16(A, B, C) \
    __builtin_amdgcn_wmma_f32_16x16x32_f16(false, (A), false, (B), (short)0, (C), false, false)

// Low 32 bits of the flat address of a __shared__ object == LDS byte offset
// (LDS aperture maps flat[31:0] -> LDS offset).
__device__ __forceinline__ unsigned lds_offset(const void* p) {
    return (unsigned)(unsigned long long)p;
}

// Per-lane async 16-byte copy global -> LDS (ASYNCcnt).
// Builtin expects (global v4i*, shared v4i*, imm offset, imm cpol).
__device__ __forceinline__ void async_copy_b128(const void* g, const void* l) {
#if defined(HAVE_ASYNC)
    typedef __attribute__((address_space(1))) v4i* gp_t;
    typedef __attribute__((address_space(3))) v4i* lp_t;
    __builtin_amdgcn_global_load_async_to_lds_b128(
        (gp_t)(unsigned long long)g,
        (lp_t)lds_offset(l), 0, 0);
#else
    (void)g; (void)l;
#endif
}

__device__ __forceinline__ void wait_async0() {
#if defined(HAVE_ASYNC)
#if __has_builtin(__builtin_amdgcn_s_wait_asynccnt)
    __builtin_amdgcn_s_wait_asynccnt(0);
#else
    asm volatile("s_wait_asynccnt 0" ::: "memory");
#endif
#endif
}

// TDM: DMA a 2D tile (tile0 elements/row x tile1 rows, f16, row stride in
// elements) from global memory into LDS. One issue per wave; TENSORcnt.
__device__ __forceinline__ void tdm_load_tile_2d(unsigned long long gaddr,
                                                 unsigned lds_off,
                                                 unsigned tile0, unsigned tile1,
                                                 unsigned stride_elems) {
#if defined(HAVE_TDM)
    // D# group 0: count=1 | lds_addr | global_addr[56:0] | type=2
    v4u g0 = { 1u,
               lds_off,
               (unsigned)(gaddr & 0xffffffffull),
               (unsigned)((gaddr >> 32) & 0x1ffffffull) | 0x80000000u };
    // D# group 1: data_size=1 (2B); big tensor dims (no OOB clip);
    // tile_dim0/1; tensor_dim0_stride = row stride.
    v8i g1 = { (int)0x00010000u,                       // wg_mask=0, data_size=1
               (int)0xffff0000u,                       // tensor_dim0 lo16
               (int)0xffff7fffu,                       // dim0 hi16 | dim1 lo16
               (int)(0x7fffu | (tile0 << 16)),         // dim1 hi16 | tile_dim0
               (int)(tile1 & 0xffffu),                 // tile_dim1 | tile_dim2=0
               (int)stride_elems,                      // dim0_stride[31:0]
               0, 0 };
    v4i gz = { 0, 0, 0, 0 };
#if defined(__clang_major__) && __clang_major__ >= 23
    v8i gz8 = { 0, 0, 0, 0, 0, 0, 0, 0 };
    __builtin_amdgcn_tensor_load_to_lds(g0, g1, gz, gz, gz8, 0);
#else
    __builtin_amdgcn_tensor_load_to_lds(g0, g1, gz, gz, 0);
#endif
#else
    (void)gaddr; (void)lds_off; (void)tile0; (void)tile1; (void)stride_elems;
#endif
}

// Problem constants (B=8, C=384, H=W=112, ws=7, heads=12)
static constexpr int DIMC   = 384;
static constexpr int NHEAD  = 12;
static constexpr int HIDD   = 1536;
static constexpr int IMGH   = 112;
static constexpr int HWSZ   = 12544;      // 112*112
static constexpr int MROWS  = 100352;     // 8*112*112 (divisible by 128)
static constexpr int NWIN   = 2048;       // 8 * 16 * 16 windows
static constexpr int NTOK   = 49;         // 7*7 tokens per window

// ---------------------------------------------------------------------------
// fp32 -> f16 elementwise convert (weights)
// ---------------------------------------------------------------------------
__global__ void cvt_f16_kernel(const float* __restrict__ in, h16* __restrict__ out, int n) {
    int i = blockIdx.x * 256 + threadIdx.x;
    if (i < n) out[i] = (h16)in[i];
}

// ---------------------------------------------------------------------------
// LayerNorm over C=384. One wave per row (wave32; 12 elements per lane).
// WINDOWED=1: read NCHW x, emit rows in window-partition order.
// WINDOWED=0: read BHWC fp32 rows.
// ---------------------------------------------------------------------------
template<int WINDOWED>
__global__ __launch_bounds__(256)
void ln_kernel(const float* __restrict__ in, const float* __restrict__ g,
               const float* __restrict__ bt, h16* __restrict__ out) {
    const int wave = threadIdx.x >> 5;
    const int lane = threadIdx.x & 31;
    const int r = blockIdx.x * 8 + wave;   // row index, < MROWS

    float vals[12];
    if (WINDOWED) {
        const int win = r / NTOK, n = r % NTOK;
        const int b   = win >> 8, rem = win & 255;
        const int hh  = (rem >> 4) * 7 + n / 7;
        const int ww  = (rem & 15) * 7 + n % 7;
        const float* p = in + (size_t)b * DIMC * HWSZ + hh * IMGH + ww;
        #pragma unroll
        for (int j = 0; j < 12; ++j) vals[j] = p[(size_t)(lane + 32 * j) * HWSZ];
    } else {
        const float* p = in + (size_t)r * DIMC;
        #pragma unroll
        for (int j = 0; j < 12; ++j) vals[j] = p[lane + 32 * j];
    }

    float s = 0.f;
    #pragma unroll
    for (int j = 0; j < 12; ++j) s += vals[j];
    #pragma unroll
    for (int m = 1; m < 32; m <<= 1) s += __shfl_xor(s, m, 32);
    const float mean = s * (1.f / 384.f);

    float v = 0.f;
    #pragma unroll
    for (int j = 0; j < 12; ++j) { float d = vals[j] - mean; v += d * d; }
    #pragma unroll
    for (int m = 1; m < 32; m <<= 1) v += __shfl_xor(v, m, 32);
    const float rinv = rsqrtf(v * (1.f / 384.f) + 1e-5f);

    h16* o = out + (size_t)r * DIMC;
    #pragma unroll
    for (int j = 0; j < 12; ++j) {
        int c = lane + 32 * j;
        o[c] = (h16)((vals[j] - mean) * rinv * g[c] + bt[c]);
    }
}

// ---------------------------------------------------------------------------
// Generic WMMA GEMM: C[M,N] = A[M,K](f16) * B[K,N](f16) + epilogue.
// Block: 256 threads (8 waves), tile 128x64, K-step 32 through LDS.
// A tile staged via TDM (wave 0), B tile via async-to-LDS when available.
// MODE 0: +bias -> f16 out                       (QKV)
// MODE 1: +bias +window-reverse residual(NCHW x) -> BHWC fp32   (proj)
// MODE 2: +bias, exact GELU -> f16 out           (FC1)
// MODE 3: +bias +residual(BHWC fp32) -> NCHW fp32 out           (FC2)
// ---------------------------------------------------------------------------
template<int MODE>
__global__ __launch_bounds__(256)
void gemm_wmma(const h16* __restrict__ A, const h16* __restrict__ Bw,
               const float* __restrict__ bias,
               h16* __restrict__ outh, float* __restrict__ outf,
               const float* __restrict__ resid,
               int M, int N, int K) {
    __shared__ alignas(16) h16 As[128 * 32];
    __shared__ alignas(16) h16 Bs[32 * 64];

    const int tid  = threadIdx.x;
    const int wave = tid >> 5;
    const int lane = tid & 31;
    const int rlo  = lane & 15;
    const int rhi  = lane >> 4;
    const size_t arow0 = (size_t)blockIdx.y * 128;
    const int    ncol0 = blockIdx.x * 64;

    FragC acc[4];
    #pragma unroll
    for (int t = 0; t < 4; ++t)
        #pragma unroll
        for (int e = 0; e < 8; ++e) acc[t].f[e] = 0.f;

    const int ksteps = K >> 5;
    for (int kt = 0; kt < ksteps; ++kt) {
        const int k0 = kt << 5;

        // ---- stage A tile (128 rows x 32 cols f16) ----
#if defined(HAVE_TDM)
        if (tid < 32) {   // wave 0 drives the tensor DMA
            tdm_load_tile_2d(
                (unsigned long long)(const void*)(A + arow0 * (size_t)K + k0),
                lds_offset(As), /*tile0=*/32, /*tile1=*/128, /*stride=*/(unsigned)K);
            __builtin_amdgcn_s_wait_tensorcnt(0);
        }
#elif defined(HAVE_ASYNC)
        #pragma unroll
        for (int rep = 0; rep < 2; ++rep) {
            int v   = tid + rep * 256;
            int row = v >> 2, kv = (v & 3) << 3;
            async_copy_b128(A + (arow0 + row) * (size_t)K + k0 + kv,
                            &As[row * 32 + kv]);
        }
#else
        #pragma unroll
        for (int rep = 0; rep < 2; ++rep) {
            int v   = tid + rep * 256;
            int row = v >> 2, kv = (v & 3) << 3;
            *(uint4*)&As[row * 32 + kv] =
                *(const uint4*)(A + (arow0 + row) * (size_t)K + k0 + kv);
        }
#endif

        // ---- stage B tile (32 rows x 64 cols f16) ----
        {
            int row = tid >> 3, nv = (tid & 7) << 3;
#if defined(HAVE_ASYNC)
            async_copy_b128(Bw + (size_t)(k0 + row) * N + ncol0 + nv,
                            &Bs[row * 64 + nv]);
#else
            *(uint4*)&Bs[row * 64 + nv] =
                *(const uint4*)(Bw + (size_t)(k0 + row) * N + ncol0 + nv);
#endif
        }
        // Prefetch next K tile into cache (global_prefetch_b8)
        if (kt + 1 < ksteps) {
            __builtin_prefetch(A + (arow0 + (tid >> 2)) * (size_t)K + k0 + 32, 0, 0);
            __builtin_prefetch(Bw + (size_t)(k0 + 32 + (tid >> 3)) * N + ncol0, 0, 0);
        }
        wait_async0();
        __syncthreads();

        // A fragment (16x32 f16, wave32 layout): lane -> row, K split by lane half
        FragA a;
        #pragma unroll
        for (int i = 0; i < 8; ++i) {
            int k = ((i >> 2) << 4) + (rhi << 3) + ((i & 3) << 1);
            a.u[i] = *(const unsigned*)&As[(wave * 16 + rlo) * 32 + k];
        }
        #pragma unroll
        for (int t = 0; t < 4; ++t) {
            FragA b;   // B fragment (32x16): lane -> N col, element e -> K (table order)
            #pragma unroll
            for (int e = 0; e < 16; ++e)
                b.v[e] = Bs[(((e >> 3) << 4) + (rhi << 3) + (e & 7)) * 64 + t * 16 + rlo];
            acc[t].v = WMMA_F16(a.v, b.v, acc[t].v);
        }
        __syncthreads();
    }

    // Epilogue. C layout: N = lane&15, M = i + 8*(lane>>4)
    #pragma unroll
    for (int t = 0; t < 4; ++t) {
        #pragma unroll
        for (int i = 0; i < 8; ++i) {
            const int mrow = (int)arow0 + wave * 16 + i + (rhi << 3);
            const int ncol = ncol0 + t * 16 + rlo;
            float val = acc[t].f[i] + bias[ncol];
            if (MODE == 0) {
                outh[(size_t)mrow * N + ncol] = (h16)val;
            } else if (MODE == 2) {
                val = 0.5f * val * (1.f + erff(val * 0.70710678118654752f));
                outh[(size_t)mrow * N + ncol] = (h16)val;
            } else if (MODE == 1) {
                // row is in window-partition order -> map back to (b,h,w)
                const int win = mrow / NTOK, n = mrow % NTOK;
                const int b   = win >> 8, rem = win & 255;
                const int hh  = (rem >> 4) * 7 + n / 7;
                const int ww  = (rem & 15) * 7 + n % 7;
                val += resid[(size_t)(b * DIMC + ncol) * HWSZ + hh * IMGH + ww];
                outf[((size_t)((b * IMGH + hh) * IMGH + ww)) * DIMC + ncol] = val;
            } else { // MODE == 3: row in BHWC order -> NCHW output
                val += resid[(size_t)mrow * DIMC + ncol];
                const int b  = mrow / HWSZ;
                const int hw = mrow % HWSZ;
                outf[(size_t)(b * DIMC + ncol) * HWSZ + hw] = val;
            }
        }
    }
}

// ---------------------------------------------------------------------------
// Window attention: one block (4 waves, 128 thr) per (window, head).
// S = (q * d^-1/2) k^T (49x49 padded to 64x64) + rel-pos bias, softmax, P v.
// ---------------------------------------------------------------------------
__global__ __launch_bounds__(128)
void attn_kernel(const h16* __restrict__ qkv, const float* __restrict__ rpb,
                 h16* __restrict__ outp) {
    __shared__ alignas(16) h16 qs [64 * 32];
    __shared__ alignas(16) h16 ksT[32 * 64];
    __shared__ alignas(16) h16 vs [64 * 32];
    __shared__ alignas(16) h16 Ps [64 * 64];

    const int head = blockIdx.x;
    const int win  = blockIdx.y;
    const int tid  = threadIdx.x;
    const float scale = 0.17677669529663687f;   // 32^-0.5

    const size_t base = (size_t)win * NTOK * 1152 + head * 32;
    for (int idx = tid; idx < 64 * 32; idx += 128) {
        const int n = idx >> 5, d = idx & 31;
        float qv = 0.f, kv = 0.f, vv = 0.f;
        if (n < NTOK) {
            const h16* p = qkv + base + (size_t)n * 1152 + d;
            qv = (float)p[0]; kv = (float)p[384]; vv = (float)p[768];
        }
        qs [n * 32 + d] = (h16)(qv * scale);
        ksT[d * 64 + n] = (h16)kv;      // k^T: (d, n) row-major, ldb=64
        vs [n * 32 + d] = (h16)vv;
    }
    __syncthreads();

    const int wave = tid >> 5, lane = tid & 31;
    const int rlo = lane & 15, rhi = lane >> 4;
    const int m0 = wave * 16 + (rhi << 3);

    // S = q @ k^T : wave owns rows [16*wave, 16*wave+16), 4 n-tiles
    FragA a;
    #pragma unroll
    for (int i = 0; i < 8; ++i) {
        int k = ((i >> 2) << 4) + (rhi << 3) + ((i & 3) << 1);
        a.u[i] = *(const unsigned*)&qs[(wave * 16 + rlo) * 32 + k];
    }
    FragC s[4];
    #pragma unroll
    for (int t = 0; t < 4; ++t) {
        FragA b;
        #pragma unroll
        for (int e = 0; e < 16; ++e)
            b.v[e] = ksT[(((e >> 3) << 4) + (rhi << 3) + (e & 7)) * 64 + t * 16 + rlo];
        s[t].v = WMMA_F16(a.v, b.v, zero8());
    }

    // relative-position bias + padding mask
    #pragma unroll
    for (int i = 0; i < 8; ++i) {
        const int m = m0 + i;
        #pragma unroll
        for (int t = 0; t < 4; ++t) {
            const int n = t * 16 + rlo;
            float v = s[t].f[i];
            if (m < NTOK && n < NTOK) {
                const int rel = ((m / 7) - (n / 7) + 6) * 13 + ((m % 7) - (n % 7) + 6);
                v += rpb[rel * NHEAD + head];
            } else {
                v = -1e30f;
            }
            s[t].f[i] = v;
        }
    }

    // row softmax: row lives in one 16-lane half across 4 tiles
    #pragma unroll
    for (int i = 0; i < 8; ++i) {
        float mx = fmaxf(fmaxf(s[0].f[i], s[1].f[i]), fmaxf(s[2].f[i], s[3].f[i]));
        mx = fmaxf(mx, __shfl_xor(mx, 1, 32));
        mx = fmaxf(mx, __shfl_xor(mx, 2, 32));
        mx = fmaxf(mx, __shfl_xor(mx, 4, 32));
        mx = fmaxf(mx, __shfl_xor(mx, 8, 32));
        float sum = 0.f;
        #pragma unroll
        for (int t = 0; t < 4; ++t) {
            float e = __expf(s[t].f[i] - mx);
            s[t].f[i] = e; sum += e;
        }
        sum += __shfl_xor(sum, 1, 32);
        sum += __shfl_xor(sum, 2, 32);
        sum += __shfl_xor(sum, 4, 32);
        sum += __shfl_xor(sum, 8, 32);
        const float r = 1.f / fmaxf(sum, 1e-20f);
        const int m = m0 + i;
        #pragma unroll
        for (int t = 0; t < 4; ++t)
            Ps[m * 64 + t * 16 + rlo] = (h16)(s[t].f[i] * r);
    }
    __syncthreads();

    // out = P @ v : (64x64)x(64x32), K loop of 2
    FragC o[2];
    #pragma unroll
    for (int t = 0; t < 2; ++t) o[t].v = zero8();
    #pragma unroll
    for (int kk = 0; kk < 64; kk += 32) {
        FragA a2;
        #pragma unroll
        for (int i = 0; i < 8; ++i) {
            int k = kk + ((i >> 2) << 4) + (rhi << 3) + ((i & 3) << 1);
            a2.u[i] = *(const unsigned*)&Ps[(wave * 16 + rlo) * 64 + k];
        }
        #pragma unroll
        for (int t = 0; t < 2; ++t) {
            FragA b;
            #pragma unroll
            for (int e = 0; e < 16; ++e)
                b.v[e] = vs[(kk + ((e >> 3) << 4) + (rhi << 3) + (e & 7)) * 32 + t * 16 + rlo];
            o[t].v = WMMA_F16(a2.v, b.v, o[t].v);
        }
    }
    #pragma unroll
    for (int i = 0; i < 8; ++i) {
        const int m = m0 + i;
        if (m < NTOK) {
            const size_t ob = ((size_t)win * NTOK + m) * DIMC + head * 32;
            #pragma unroll
            for (int t = 0; t < 2; ++t)
                outp[ob + t * 16 + rlo] = (h16)o[t].f[i];
        }
    }
}

// ---------------------------------------------------------------------------
// Host driver
// ---------------------------------------------------------------------------
extern "C" void kernel_launch(void* const* d_in, const int* in_sizes, int n_in,
                              void* d_out, int out_size, void* d_ws, size_t ws_size,
                              hipStream_t stream) {
    (void)in_sizes; (void)n_in; (void)out_size; (void)ws_size;
    const float* x      = (const float*)d_in[0];
    const float* g1     = (const float*)d_in[1];
    const float* b1     = (const float*)d_in[2];
    const float* w_qkv  = (const float*)d_in[3];
    const float* b_qkv  = (const float*)d_in[4];
    const float* rpb    = (const float*)d_in[5];
    const float* w_proj = (const float*)d_in[6];
    const float* b_proj = (const float*)d_in[7];
    const float* g2     = (const float*)d_in[8];
    const float* b2     = (const float*)d_in[9];
    const float* w_fc1  = (const float*)d_in[10];
    const float* b_fc1  = (const float*)d_in[11];
    const float* w_fc2  = (const float*)d_in[12];
    const float* b_fc2  = (const float*)d_in[13];
    float* out = (float*)d_out;

    // Workspace layout (bytes), 256-aligned, with aliasing:
    //   [0)        xh / hh : MROWS*384 f16
    //   [OFF_QKV)  qkvh    : MROWS*1152 f16 -- later reused (+attn region)
    //              as acth : MROWS*1536 f16
    //   [OFF_ATT)  attnh   : MROWS*384 f16
    //   [OFF_X2)   x2      : MROWS*384 f32
    //   [OFF_W*)   f16 weights
    char* ws = (char*)d_ws;
    constexpr size_t SZ_XH   = (size_t)MROWS * DIMC * 2;
    constexpr size_t SZ_QKV  = (size_t)MROWS * 1152 * 2;
    constexpr size_t SZ_ATT  = (size_t)MROWS * DIMC * 2;
    constexpr size_t SZ_X2   = (size_t)MROWS * DIMC * 4;
    constexpr size_t OFF_QKV = SZ_XH;
    constexpr size_t OFF_ATT = OFF_QKV + SZ_QKV;
    constexpr size_t OFF_X2  = OFF_ATT + SZ_ATT;
    constexpr size_t OFF_WQ  = OFF_X2 + SZ_X2;
    constexpr size_t OFF_WP  = OFF_WQ + (size_t)DIMC * 1152 * 2;
    constexpr size_t OFF_W1  = OFF_WP + (size_t)DIMC * DIMC * 2;
    constexpr size_t OFF_W2  = OFF_W1 + (size_t)DIMC * HIDD * 2;

    h16*   xh    = (h16*)(ws);             // LN1 output / LN2 output (reuse)
    h16*   qkvh  = (h16*)(ws + OFF_QKV);
    h16*   acth  = (h16*)(ws + OFF_QKV);   // aliases dead qkv+attn regions
    h16*   attnh = (h16*)(ws + OFF_ATT);
    float* x2    = (float*)(ws + OFF_X2);
    h16*   wqh   = (h16*)(ws + OFF_WQ);
    h16*   wph   = (h16*)(ws + OFF_WP);
    h16*   w1h   = (h16*)(ws + OFF_W1);
    h16*   w2h   = (h16*)(ws + OFF_W2);

    // 1) weight conversion fp32 -> f16
    cvt_f16_kernel<<<(DIMC * 1152 + 255) / 256, 256, 0, stream>>>(w_qkv, wqh, DIMC * 1152);
    cvt_f16_kernel<<<(DIMC * DIMC + 255) / 256, 256, 0, stream>>>(w_proj, wph, DIMC * DIMC);
    cvt_f16_kernel<<<(DIMC * HIDD + 255) / 256, 256, 0, stream>>>(w_fc1, w1h, DIMC * HIDD);
    cvt_f16_kernel<<<(HIDD * DIMC + 255) / 256, 256, 0, stream>>>(w_fc2, w2h, HIDD * DIMC);

    // 2) LN1 + window partition (NCHW -> windowed rows, f16)
    ln_kernel<1><<<MROWS / 8, 256, 0, stream>>>(x, g1, b1, xh);

    // 3) QKV GEMM: (100352 x 384) x (384 x 1152)
    gemm_wmma<0><<<dim3(1152 / 64, MROWS / 128), 256, 0, stream>>>(
        xh, wqh, b_qkv, qkvh, nullptr, nullptr, MROWS, 1152, DIMC);

    // 4) window attention per (head, window)
    attn_kernel<<<dim3(NHEAD, NWIN), 128, 0, stream>>>(qkvh, rpb, attnh);

    // 5) proj GEMM + window reverse + residual -> x2 (BHWC fp32)
    gemm_wmma<1><<<dim3(DIMC / 64, MROWS / 128), 256, 0, stream>>>(
        attnh, wph, b_proj, nullptr, x2, x, MROWS, DIMC, DIMC);

    // 6) LN2 on x2 -> hh (f16, BHWC row order)
    ln_kernel<0><<<MROWS / 8, 256, 0, stream>>>(x2, g2, b2, xh);

    // 7) FC1 + GELU: (100352 x 384) x (384 x 1536)
    gemm_wmma<2><<<dim3(HIDD / 64, MROWS / 128), 256, 0, stream>>>(
        xh, w1h, b_fc1, acth, nullptr, nullptr, MROWS, HIDD, DIMC);

    // 8) FC2 + residual -> final NCHW fp32 output
    gemm_wmma<3><<<dim3(DIMC / 64, MROWS / 128), 256, 0, stream>>>(
        acth, w2h, b_fc2, nullptr, out, x2, MROWS, DIMC, HIDD);
}
